// NexusEmergence_15676630630960
// MI455X (gfx1250) — matrix operand order repaired and checked by reading, not compiled, not run on hardware
//
#include <hip/hip_runtime.h>
#include <hip/hip_bf16.h>
#include <math.h>

// ---------------------------------------------------------------------------
// Types for CDNA5 WMMA (wave32): v_wmma_f32_16x16x32_bf16
// ---------------------------------------------------------------------------
typedef __bf16 bf16_t;
typedef __attribute__((ext_vector_type(16))) __bf16 v16bf;
typedef __attribute__((ext_vector_type(8)))  float  v8f;

union FragBF {
    v16bf h;
    uint4 u[2];
};
union Pack8 {
    bf16_t h[8];
    uint4  u;
};

// ---------------------------------------------------------------------------
// Unified implicit-im2col WMMA GEMM, compile-time specialized.
//   D[M][N] = A[M][K] * B[K][N] + bias  (opt relu, opt per-batch scale)
//   DIN==false : B = NCHW activations, k = c*(R*S)+r*S+s, zero pad
//                n_abs = n_begin + n ; b = n_abs>>12 ; pos = n_abs&4095
//   DIN==true  : B = compact [K][N] chunk (FFN hidden)
//   DOUT==false: NCHW out[((b*M)+m)*4096+pos] ; DOUT==true: compact [M][N]
//
// INVARIANTS (guaranteed by all call sites; lets us drop ALL bounds checks):
//   M % (MWAVES*32) == 0,  K % 32 == 0,  N % 128 == 0
//
// 256 threads = 8 wave32s. BN = 128 always.
//   MWAVES=4: BM=128, waves 4x2, wave tile 32x64 (2x4 accs, 8 wmma / K-step)
//   MWAVES=2: BM= 64, waves 2x4, wave tile 32x32 (2x2 accs, 4 wmma / K-step)
// Register-pipelined staging: next K-tile global loads issue before the WMMAs.
// LDS row stride 56 bf16 (112B): 16B-aligned ds_load_b128, conflict-free.
// ---------------------------------------------------------------------------
template<int MWAVES, bool DIN, bool DOUT, bool RELU, bool BSC>
__global__ __launch_bounds__(256)
void gemm_wmma_kernel(const float* __restrict__ A,
                      const float* __restrict__ Bx,
                      const float* __restrict__ bias,
                      float* __restrict__ Out,
                      const float* __restrict__ bscale,
                      int M, int K, int N, int n_begin,
                      int Cin, int R, int S, int pad,
                      float in_scale)
{
    constexpr int BM     = MWAVES * 32;
    constexpr int BN     = 128;
    constexpr int NWAVES = 8 / MWAVES;
    constexpr int WN     = BN / NWAVES;   // wave N span: 64 or 32
    constexpr int NT     = WN / 16;       // 4 or 2
    constexpr int LDSB   = 56;            // padded bf16 row stride (112B)
    constexpr int TPR    = 256 / BM;      // threads per A row
    constexpr int KPT    = 32 / TPR;      // k per thread (A stage): 16 or 8

    __shared__ __align__(16) bf16_t As[BM * LDSB];   // [m][k]
    __shared__ __align__(16) bf16_t Bs[BN * LDSB];   // [n][k]

    const int tid  = threadIdx.x;
    const int lane = tid & 31;
    const int wid  = tid >> 5;
    const int lm   = lane & 15;
    const int lh   = lane >> 4;
    const int m0   = blockIdx.y * BM;
    const int n0   = blockIdx.x * BN;
    const int mrow = (wid % MWAVES) * 32;
    const int ncol = (wid / MWAVES) * WN;
    const int rs_sz = R * S;

    // staging assignments
    const int arow  = tid / TPR;
    const int akoff = (tid % TPR) * KPT;
    const int amg   = m0 + arow;
    const int bnl   = tid & 127;
    const int bkoff = (tid >> 7) * 16;
    const int bng   = n0 + bnl;

    float ra[KPT];
    float rb[16];

    // A: contiguous in k, offsets multiples of 16B by the invariants -> b128s
    auto load_a = [&](int k0) {
        const float* p = &A[(size_t)amg * K + k0 + akoff];
#pragma unroll
        for (int i = 0; i < KPT; i += 4) {
            float4 t = *(const float4*)(p + i);
            ra[i + 0] = t.x; ra[i + 1] = t.y; ra[i + 2] = t.z; ra[i + 3] = t.w;
        }
    };
    auto load_b = [&](int k0) {
#pragma unroll
        for (int i = 0; i < 16; ++i) {
            int kg = k0 + bkoff + i;
            if (!DIN) {
                int n_abs = n_begin + bng;
                int b = n_abs >> 12, pos = n_abs & 4095;
                int c = kg / rs_sz, rs = kg % rs_sz;
                int r = rs / S, s = rs % S;
                int y = (pos >> 6) + r - pad;
                int x = (pos & 63) + s - pad;
                float v = 0.0f;
                if ((unsigned)y < 64u && (unsigned)x < 64u)
                    v = Bx[((size_t)b * Cin + c) * 4096 + y * 64 + x] * in_scale;
                rb[i] = v;
            } else {
                rb[i] = Bx[(size_t)kg * N + bng];
            }
        }
    };

    v8f acc[2][NT];
#pragma unroll
    for (int mt = 0; mt < 2; ++mt)
#pragma unroll
        for (int nt = 0; nt < NT; ++nt) acc[mt][nt] = v8f{};

    load_a(0);
    load_b(0);

    for (int k0 = 0; k0 < K; k0 += 32) {
        // ---- stage regs -> LDS (fp32 -> bf16, vector stores) ----
#pragma unroll
        for (int i = 0; i < KPT; i += 8) {
            Pack8 t;
#pragma unroll
            for (int j = 0; j < 8; ++j) t.h[j] = (bf16_t)ra[i + j];
            *(uint4*)&As[arow * LDSB + akoff + i] = t.u;
        }
#pragma unroll
        for (int i = 0; i < 16; i += 8) {
            Pack8 t;
#pragma unroll
            for (int j = 0; j < 8; ++j) t.h[j] = (bf16_t)rb[i + j];
            *(uint4*)&Bs[bnl * LDSB + bkoff + i] = t.u;
        }
        __syncthreads();

        // ---- issue next tile's global loads (hidden under the WMMAs) ----
        if (k0 + 32 < K) {
            load_a(k0 + 32);
            load_b(k0 + 32);
        }

        // ---- fragments (ISA 16-bit A 16x32 / B 32x16 layouts) + WMMA ----
        FragBF af[2], bf[NT];
#pragma unroll
        for (int mt = 0; mt < 2; ++mt) {
            const int row = (mrow + mt * 16 + lm) * LDSB;
            af[mt].u[0] = *(const uint4*)&As[row + lh * 8];
            af[mt].u[1] = *(const uint4*)&As[row + 16 + lh * 8];
        }
#pragma unroll
        for (int nt = 0; nt < NT; ++nt) {
            const int row = (ncol + nt * 16 + lm) * LDSB;
            bf[nt].u[0] = *(const uint4*)&Bs[row + lh * 8];
            bf[nt].u[1] = *(const uint4*)&Bs[row + 16 + lh * 8];
        }
#pragma unroll
        for (int mt = 0; mt < 2; ++mt)
#pragma unroll
            for (int nt = 0; nt < NT; ++nt)
                acc[mt][nt] = __builtin_amdgcn_wmma_f32_16x16x32_bf16(
                    false, af[mt].h, false, bf[nt].h, (short)0, acc[mt][nt], false, false);
        __syncthreads();
    }

    // ---- epilogue: hoisted bias, straight-line stores (no bounds checks) ----
    float bv[2][8];
#pragma unroll
    for (int mt = 0; mt < 2; ++mt)
#pragma unroll
        for (int r = 0; r < 8; ++r)
            bv[mt][r] = bias[m0 + mrow + mt * 16 + lh * 8 + r];

#pragma unroll
    for (int mt = 0; mt < 2; ++mt) {
#pragma unroll
        for (int nt = 0; nt < NT; ++nt) {
            int ng = n0 + ncol + nt * 16 + lm;
            int n_abs = n_begin + ng;
            int b = n_abs >> 12, pos = n_abs & 4095;
            float sc = 1.0f;
            if (BSC) sc = bscale[b];
#pragma unroll
            for (int r = 0; r < 8; ++r) {
                int mg = m0 + mrow + mt * 16 + lh * 8 + r;
                float val = acc[mt][nt][r] + bv[mt][r];
                if (RELU) val = fmaxf(val, 0.0f);
                if (!DOUT) {
                    if (BSC) val *= sc;
                    Out[((size_t)b * M + mg) * 4096 + pos] = val;
                } else {
                    Out[(size_t)mg * N + ng] = val;
                }
            }
        }
    }
}

template<int MW, bool DIN, bool DOUT, bool RELU, bool BSC>
static inline void run_gemm(hipStream_t stream,
                            const float* A, const float* Bx, const float* bias, float* Out,
                            const float* bscale, int M, int K, int N, int n_begin,
                            int Cin, int R, int S, int pad, float in_scale)
{
    // invariants: M % (MW*32) == 0, K % 32 == 0, N % 128 == 0
    dim3 grid(N / 128, M / (MW * 32));
    gemm_wmma_kernel<MW, DIN, DOUT, RELU, BSC><<<grid, 256, 0, stream>>>(
        A, Bx, bias, Out, bscale, M, K, N, n_begin, Cin, R, S, pad, in_scale);
}

// ---------------------------------------------------------------------------
// Gaussian blur (separable, per-(b,c) sigma, truncate radius, symmetric pad)
// ---------------------------------------------------------------------------
__global__ void blur_weights_kernel(const float* __restrict__ u, float* __restrict__ w)
{
    int i = blockIdx.x * blockDim.x + threadIdx.x;
    if (i >= 2048) return;
    float sig = 0.5f + 1.5f * u[i];
    float radius = floorf(4.0f * sig + 0.5f);
    float tap[17];
    float acc = 0.0f;
#pragma unroll
    for (int k = 0; k < 17; ++k) {
        float off = (float)(k - 8);
        float t = (fabsf(off) <= radius) ? __expf(-0.5f * (off / sig) * (off / sig)) : 0.0f;
        tap[k] = t;
        acc += t;
    }
    float inv = 1.0f / acc;
#pragma unroll
    for (int k = 0; k < 17; ++k) w[i * 17 + k] = tap[k] * inv;
}

__device__ __forceinline__ int symref(int i)
{
    if (i < 0)   return -1 - i;
    if (i >= 64) return 127 - i;
    return i;
}

__global__ void blur_pass_kernel(const float* __restrict__ in, const float* __restrict__ w,
                                 float* __restrict__ out, int horiz)
{
    int idx = blockIdx.x * blockDim.x + threadIdx.x;
    if (idx >= 4 * 512 * 4096) return;
    int pos = idx & 4095, bc = idx >> 12;
    int y = pos >> 6, x = pos & 63;
    const float* wp   = w + bc * 17;
    const float* base = in + (size_t)bc * 4096;
    float acc = 0.0f;
#pragma unroll
    for (int k = 0; k < 17; ++k) {
        int yy = y, xx = x;
        if (horiz) xx = symref(x - 8 + k);
        else       yy = symref(y - 8 + k);
        acc += wp[k] * base[yy * 64 + xx];
    }
    out[idx] = acc;
}

// ---------------------------------------------------------------------------
// Per-(b,c) spatial mean; align head softmax -> per-batch scale
// ---------------------------------------------------------------------------
__global__ void mean_kernel(const float* __restrict__ in, float* __restrict__ out)
{
    __shared__ float red[256];
    int bc = blockIdx.x;
    const float* p = in + (size_t)bc * 4096;
    float s = 0.0f;
    for (int i = threadIdx.x; i < 4096; i += 256) s += p[i];
    red[threadIdx.x] = s;
    __syncthreads();
    for (int off = 128; off > 0; off >>= 1) {
        if (threadIdx.x < off) red[threadIdx.x] += red[threadIdx.x + off];
        __syncthreads();
    }
    if (threadIdx.x == 0) out[bc] = red[0] * (1.0f / 4096.0f);
}

__global__ void align_scale_kernel(const float* __restrict__ mean, const float* __restrict__ aw,
                                   const float* __restrict__ ab, float* __restrict__ scale)
{
    __shared__ float z[16];
    int t = threadIdx.x;
    if (t < 16) {
        int b = t >> 2, j = t & 3;
        float acc = ab[j];
        for (int c = 0; c < 512; ++c) acc += mean[b * 512 + c] * aw[j * 512 + c];
        z[t] = acc;
    }
    __syncthreads();
    if (t < 4) {
        float m = z[t * 4];
        for (int j = 1; j < 4; ++j) m = fmaxf(m, z[t * 4 + j]);
        float s = 0.0f;
        for (int j = 0; j < 4; ++j) s += __expf(z[t * 4 + j] - m);
        float wsum = 0.0f;
        for (int j = 0; j < 4; ++j) wsum += __expf(z[t * 4 + j] - m) / s;
        scale[t] = 1.0f + 0.1f * wsum;
    }
}

// ---------------------------------------------------------------------------
// MHA core: sequence length 4 (over B), batch = HW positions, 16 heads, dh=4
// qkv NCHW [l][192][pos]: q=ch 0..63, k=64..127, v=128..191
// ---------------------------------------------------------------------------
__global__ void attention_kernel(const float* __restrict__ qkv, float* __restrict__ o)
{
    int idx = blockIdx.x * blockDim.x + threadIdx.x;
    if (idx >= 4096 * 16) return;
    int pos = idx & 4095, h = idx >> 12;

    float q[4][4], kk[4][4], v[4][4];
#pragma unroll
    for (int l = 0; l < 4; ++l) {
#pragma unroll
        for (int d = 0; d < 4; ++d) {
            size_t base = (size_t)l * 192 * 4096 + pos;
            int ch = h * 4 + d;
            q[l][d]  = qkv[base + (size_t)ch * 4096];
            kk[l][d] = qkv[base + (size_t)(64 + ch) * 4096];
            v[l][d]  = qkv[base + (size_t)(128 + ch) * 4096];
        }
    }
#pragma unroll
    for (int l = 0; l < 4; ++l) {
        float sc[4], m = -1e30f;
#pragma unroll
        for (int mm = 0; mm < 4; ++mm) {
            float s = 0.0f;
#pragma unroll
            for (int d = 0; d < 4; ++d) s += q[l][d] * kk[mm][d];
            sc[mm] = 0.5f * s;                      // q * dh^-0.5, dh=4
            m = fmaxf(m, sc[mm]);
        }
        float denom = 0.0f;
#pragma unroll
        for (int mm = 0; mm < 4; ++mm) { sc[mm] = __expf(sc[mm] - m); denom += sc[mm]; }
        float inv = 1.0f / denom;
        float od[4] = {0.f, 0.f, 0.f, 0.f};
#pragma unroll
        for (int mm = 0; mm < 4; ++mm)
#pragma unroll
            for (int d = 0; d < 4; ++d) od[d] += sc[mm] * inv * v[mm][d];
#pragma unroll
        for (int d = 0; d < 4; ++d)
            o[(size_t)l * 64 * 4096 + (size_t)(h * 4 + d) * 4096 + pos] = od[d];
    }
}

// LayerNorm over 64 channels of (a + b), NCHW layout
__global__ void ln_kernel(const float* __restrict__ a, const float* __restrict__ b,
                          const float* __restrict__ g, const float* __restrict__ beta,
                          float* __restrict__ out)
{
    int idx = blockIdx.x * blockDim.x + threadIdx.x;
    if (idx >= 16384) return;
    int bb = idx >> 12, pos = idx & 4095;
    size_t base = (size_t)bb * 64 * 4096 + pos;
    float xs[64];
    float mean = 0.0f;
#pragma unroll
    for (int c = 0; c < 64; ++c) {
        float v = a[base + (size_t)c * 4096] + b[base + (size_t)c * 4096];
        xs[c] = v;
        mean += v;
    }
    mean *= (1.0f / 64.0f);
    float var = 0.0f;
#pragma unroll
    for (int c = 0; c < 64; ++c) { float d = xs[c] - mean; var += d * d; }
    var *= (1.0f / 64.0f);
    float inv = rsqrtf(var + 1e-5f);
#pragma unroll
    for (int c = 0; c < 64; ++c)
        out[base + (size_t)c * 4096] = (xs[c] - mean) * inv * g[c] + beta[c];
}

__global__ void copy_kernel(const float* __restrict__ in, float* __restrict__ out, int n)
{
    int i = blockIdx.x * blockDim.x + threadIdx.x;
    if (i < n) out[i] = in[i];
}

__global__ void add_kernel(const float* __restrict__ in, float* __restrict__ out, int n)
{
    int i = blockIdx.x * blockDim.x + threadIdx.x;
    if (i < n) out[i] += in[i];
}

// out = conv1x1(combined, out_w[1x64]) + out_b
__global__ void outconv_kernel(const float* __restrict__ comb, const float* __restrict__ w,
                               const float* __restrict__ b, float* __restrict__ out)
{
    int idx = blockIdx.x * blockDim.x + threadIdx.x;
    if (idx >= 16384) return;
    int bb = idx >> 12, pos = idx & 4095;
    float acc = b[0];
#pragma unroll
    for (int c = 0; c < 64; ++c)
        acc += comb[(size_t)bb * 64 * 4096 + (size_t)c * 4096 + pos] * w[c];
    out[idx] = acc;
}

// ---------------------------------------------------------------------------
// Host orchestration
// ---------------------------------------------------------------------------
extern "C" void kernel_launch(void* const* d_in, const int* in_sizes, int n_in,
                              void* d_out, int out_size, void* d_ws, size_t ws_size,
                              hipStream_t stream)
{
    (void)in_sizes; (void)n_in; (void)out_size; (void)ws_size;

    const float* x       = (const float*)d_in[0];
    const float* sigma_u = (const float*)d_in[1];
    const float *bw1[4], *bb1[4], *bw2[4], *bb2[4];
    for (int i = 0; i < 4; ++i) {
        bw1[i] = (const float*)d_in[2 + 4 * i];
        bb1[i] = (const float*)d_in[3 + 4 * i];
        bw2[i] = (const float*)d_in[4 + 4 * i];
        bb2[i] = (const float*)d_in[5 + 4 * i];
    }
    const float* align_w = (const float*)d_in[18];
    const float* align_b = (const float*)d_in[19];
    const float* final_w = (const float*)d_in[20];
    const float* final_b = (const float*)d_in[21];
    struct GenP { const float *wqkv, *bqkv, *wo, *bo, *g1, *be1, *w1, *bl1, *w2, *bl2, *g2, *be2; } gens[4];
    for (int g = 0; g < 4; ++g) {
        int p = 22 + 12 * g;
        gens[g].wqkv = (const float*)d_in[p + 0];  gens[g].bqkv = (const float*)d_in[p + 1];
        gens[g].wo   = (const float*)d_in[p + 2];  gens[g].bo   = (const float*)d_in[p + 3];
        gens[g].g1   = (const float*)d_in[p + 4];  gens[g].be1  = (const float*)d_in[p + 5];
        gens[g].w1   = (const float*)d_in[p + 6];  gens[g].bl1  = (const float*)d_in[p + 7];
        gens[g].w2   = (const float*)d_in[p + 8];  gens[g].bl2  = (const float*)d_in[p + 9];
        gens[g].g2   = (const float*)d_in[p + 10]; gens[g].be2  = (const float*)d_in[p + 11];
    }
    const float* out_w = (const float*)d_in[70];
    const float* out_b = (const float*)d_in[71];

    // workspace layout (floats)
    float* ws     = (float*)d_ws;
    float* buf0   = ws;                      // 512*16384
    float* buf1   = buf0 + 8388608;          // 512*16384
    float* sbuf   = buf1 + 8388608;          // 64*16384
    float* qkvb   = sbuf + 1048576;          // 192*16384
    float* attn   = qkvb + 3145728;          // 64*16384
    float* ta     = attn + 1048576;          // 64*16384
    float* tb     = ta   + 1048576;          // 64*16384
    float* tc     = tb   + 1048576;          // 64*16384
    float* h1     = tc   + 1048576;          // 2048*2048 FFN chunk
    float* bwts   = h1   + 4194304;          // 2048*17 blur taps
    float* meanb  = bwts + 2048 * 17;        // 2048
    float* scaleb = meanb + 2048;            // 4

    float* outp = (float*)d_out;             // (4,1,64,64)  = 16384
    float* comb = outp + 16384;              // (4,64,64,64) = 1048576

    const int N = 16384;

    // ---- substrate conv stack (implicit-im2col bf16 WMMA) ----
    run_gemm<4, false, false, true,  false>(stream, bw1[0], x,    bb1[0], buf0, nullptr, 512,  64 * 9, N, 0,  64, 3, 3, 1, 1.0f);
    run_gemm<4, false, false, false, false>(stream, bw2[0], buf0, bb2[0], buf1, nullptr, 512, 512 * 9, N, 0, 512, 3, 3, 1, 1.0f);
    for (int i = 1; i < 4; ++i) {
        run_gemm<4, false, false, true,  false>(stream, bw1[i], buf1, bb1[i], buf0, nullptr, 512, 512 * 9, N, 0, 512, 3, 3, 1, 2.0f);
        run_gemm<4, false, false, false, false>(stream, bw2[i], buf0, bb2[i], buf1, nullptr, 512, 512 * 9, N, 0, 512, 3, 3, 1, 1.0f);
    }

    // ---- per-(b,c) gaussian blur, vertical then horizontal ----
    blur_weights_kernel<<<16, 128, 0, stream>>>(sigma_u, bwts);
    blur_pass_kernel<<<(4 * 512 * 4096) / 256, 256, 0, stream>>>(buf1, bwts, buf0, 0);
    blur_pass_kernel<<<(4 * 512 * 4096) / 256, 256, 0, stream>>>(buf0, bwts, buf1, 1);

    // ---- alignment softmax -> per-batch scale; final 1x1 conv (fused scale) ----
    mean_kernel<<<2048, 256, 0, stream>>>(buf1, meanb);
    align_scale_kernel<<<1, 32, 0, stream>>>(meanb, align_w, align_b, scaleb);
    run_gemm<2, false, false, false, true>(stream, final_w, buf1, final_b, sbuf, scaleb, 64, 512, N, 0, 512, 1, 1, 0, 1.0f);

    // combined starts as s (written straight into d_out)
    copy_kernel<<<1048576 / 256, 256, 0, stream>>>(sbuf, comb, 1048576);

    // ---- 4 pattern generators (all linear layers via the same WMMA GEMM) ----
    for (int g = 0; g < 4; ++g) {
        run_gemm<2, false, false, false, false>(stream, gens[g].wqkv, sbuf, gens[g].bqkv, qkvb, nullptr, 192, 64, N, 0, 64, 1, 1, 0, 1.0f);
        attention_kernel<<<65536 / 256, 256, 0, stream>>>(qkvb, attn);
        run_gemm<2, false, false, false, false>(stream, gens[g].wo, attn, gens[g].bo, ta, nullptr, 64, 64, N, 0, 64, 1, 1, 0, 1.0f);
        ln_kernel<<<16384 / 256, 256, 0, stream>>>(ta, sbuf, gens[g].g1, gens[g].be1, tb);  // x1 = LN(mha + x)
        for (int nb = 0; nb < 8; ++nb) {  // FFN chunked over N to bound workspace
            run_gemm<4, false, true,  true,  false>(stream, gens[g].w1, tb, gens[g].bl1, h1, nullptr, 2048,   64, 2048, nb * 2048, 64, 1, 1, 0, 1.0f);
            run_gemm<2, true,  false, false, false>(stream, gens[g].w2, h1, gens[g].bl2, tc, nullptr,   64, 2048, 2048, nb * 2048,  0, 1, 1, 0, 1.0f);
        }
        ln_kernel<<<16384 / 256, 256, 0, stream>>>(tc, tb, gens[g].g2, gens[g].be2, ta);    // p = LN(ffn + x1)
        add_kernel<<<1048576 / 256, 256, 0, stream>>>(ta, comb, 1048576);
    }

    // ---- output head ----
    outconv_kernel<<<16384 / 256, 256, 0, stream>>>(comb, out_w, out_b, outp);
}
